// ScaledDotProductAttention_54915451846832
// MI455X (gfx1250) — matrix operand order repaired and checked by reading, not compiled, not run on hardware
//
#include <hip/hip_runtime.h>

typedef __attribute__((ext_vector_type(16))) __bf16 v16bf;
typedef __attribute__((ext_vector_type(8)))  float  v8f;
typedef __attribute__((ext_vector_type(2)))  float  f32x2;

#define BATCH 8
#define NN    4096
#define DD    128
#define KSTRIDE 144   // padded LDS row stride (halfs): 288B rows -> 32B-aligned v16bf loads

union AF { v16bf v; __bf16 e[16]; unsigned u[8]; };

__device__ __forceinline__ float half16_max(float x) {
  #pragma unroll
  for (int m = 1; m < 16; m <<= 1) x = fmaxf(x, __shfl_xor(x, m, 32));
  return x;
}
__device__ __forceinline__ float half16_sum(float x) {
  #pragma unroll
  for (int m = 1; m < 16; m <<= 1) x += __shfl_xor(x, m, 32);
  return x;
}

__global__ __launch_bounds__(256, 1)
void attn_fused_kernel(const float* __restrict__ Q, const float* __restrict__ K,
                       const float* __restrict__ V, float* __restrict__ Out,
                       float* __restrict__ Att)
{
  extern __shared__ __attribute__((aligned(32))) __bf16 lds[];
  __bf16* ldsK = lds;                       // [128][KSTRIDE], key-major bf16
  __bf16* ldsV = lds + 128 * KSTRIDE;       // [128][KSTRIDE], d-major (transposed) bf16

  const int tid  = threadIdx.x;
  const int wave = tid >> 5;
  const int lane = tid & 31;
  const int l16  = lane & 15;
  const int hi   = lane >> 4;
  __bf16* ldsP = lds + 2 * 128 * KSTRIDE + wave * (16 * 32);  // per-wave P bounce buffer

  const int b     = blockIdx.y;
  const int qbase = blockIdx.x * 128 + wave * 16;
  const float scale = 0.08838834764831845f;  // 1/sqrt(128)

  // ---- Q A-fragments (16 rows x 128 cols -> 4 frags of 16x32, bf16), contiguous k = 16*hi + s ----
  // Q is read exactly once per block: non-temporal loads, don't pollute L2.
  AF qf[4];
  {
    const float* qrow = Q + ((size_t)b * NN + qbase + l16) * DD;
    #pragma unroll
    for (int f = 0; f < 4; ++f) {
      #pragma unroll
      for (int p = 0; p < 8; ++p) {
        int d = 32 * f + 2 * p + 16 * hi;
        f32x2 t = __builtin_nontemporal_load((const f32x2*)(qrow + d));
        qf[f].e[2 * p]     = (__bf16)t.x;
        qf[f].e[2 * p + 1] = (__bf16)t.y;
      }
    }
  }

  float mrow[8], lrow[8];
  #pragma unroll
  for (int r = 0; r < 8; ++r) { mrow[r] = -3.0e38f; lrow[r] = 0.f; }

  v8f oacc[8];
  #pragma unroll
  for (int n = 0; n < 8; ++n) oacc[n] = (v8f)(0.f);

  float invl[8];

  const int srow = tid >> 1;            // staging row 0..127
  const int sc0  = (tid & 1) * 64;      // staging col 0 or 64

  for (int pass = 0; pass < 2; ++pass) {
    if (pass == 1) {
      #pragma unroll
      for (int r = 0; r < 8; ++r) invl[r] = 1.0f / lrow[r];
    }

    for (int kc = 0; kc < NN / 128; ++kc) {
      // ---- stage K chunk (and V chunk, pass 2) into LDS as bf16 ----
      {
        const float* krow = K + ((size_t)b * NN + kc * 128 + srow) * DD + sc0;
        #pragma unroll
        for (int i = 0; i < 16; ++i) {
          float4 t = *(const float4*)(krow + 4 * i);
          __bf16* dst = ldsK + srow * KSTRIDE + sc0 + 4 * i;
          dst[0] = (__bf16)t.x; dst[1] = (__bf16)t.y;
          dst[2] = (__bf16)t.z; dst[3] = (__bf16)t.w;
        }
        if (pass == 1) {
          const float* vrow = V + ((size_t)b * NN + kc * 128 + srow) * DD + sc0;
          #pragma unroll
          for (int i = 0; i < 16; ++i) {
            float4 t = *(const float4*)(vrow + 4 * i);
            int d = sc0 + 4 * i;
            ldsV[(d + 0) * KSTRIDE + srow] = (__bf16)t.x;
            ldsV[(d + 1) * KSTRIDE + srow] = (__bf16)t.y;
            ldsV[(d + 2) * KSTRIDE + srow] = (__bf16)t.z;
            ldsV[(d + 3) * KSTRIDE + srow] = (__bf16)t.w;
          }
        }
        // prefetch next chunk's segment (this thread's 256B slice = 2 lines)
        if (kc + 1 < NN / 128) {
          const float* kn = K + ((size_t)b * NN + (kc + 1) * 128 + srow) * DD + sc0;
          __builtin_prefetch(kn, 0, 3);
          __builtin_prefetch(kn + 32, 0, 3);
          if (pass == 1) {
            const float* vn = V + ((size_t)b * NN + (kc + 1) * 128 + srow) * DD + sc0;
            __builtin_prefetch(vn, 0, 3);
            __builtin_prefetch(vn + 32, 0, 3);
          }
        }
      }
      __syncthreads();

      // ---- S tile: this wave's 16 q-rows x 128 keys (8 subtiles of 16x16) ----
      v8f sacc[8];
      #pragma unroll
      for (int j = 0; j < 8; ++j) sacc[j] = (v8f)(0.f);
      #pragma unroll
      for (int j = 0; j < 8; ++j) {
        #pragma unroll
        for (int f = 0; f < 4; ++f) {
          const v16bf bk = *(const v16bf*)(ldsK + (j * 16 + l16) * KSTRIDE + 32 * f + 16 * hi);
          sacc[j] = __builtin_amdgcn_wmma_f32_16x16x32_bf16(
              false, qf[f].v, false, bk, (short)0, sacc[j], false, false);
        }
      }

      if (pass == 0) {
        // ---- online softmax statistics (flash-style) ----
        #pragma unroll
        for (int r = 0; r < 8; ++r) {
          float x = sacc[0][r];
          #pragma unroll
          for (int j = 1; j < 8; ++j) x = fmaxf(x, sacc[j][r]);
          float cm = half16_max(x) * scale;
          float nm = fmaxf(mrow[r], cm);
          float cs = 0.f;
          #pragma unroll
          for (int j = 0; j < 8; ++j) cs += __expf(sacc[j][r] * scale - nm);
          cs = half16_sum(cs);
          lrow[r] = lrow[r] * __expf(mrow[r] - nm) + cs;
          mrow[r] = nm;
        }
      } else {
        // ---- normalized probabilities: write attention (streaming NT) + O = P @ V ----
        float* attb = Att + (size_t)b * NN * NN;
        #pragma unroll
        for (int j = 0; j < 8; ++j) {
          #pragma unroll
          for (int r = 0; r < 8; ++r) {
            float a = __expf(sacc[j][r] * scale - mrow[r]) * invl[r];
            sacc[j][r] = a;
            int q = qbase + r + 8 * hi;
            int k = kc * 128 + j * 16 + l16;
            __builtin_nontemporal_store(a, &attb[(size_t)q * NN + k]);
          }
        }
        #pragma unroll
        for (int jp = 0; jp < 4; ++jp) {
          // C-layout -> A-layout via per-wave LDS bounce (16x32 bf16)
          #pragma unroll
          for (int jj = 0; jj < 2; ++jj) {
            int j = jp * 2 + jj;
            #pragma unroll
            for (int r = 0; r < 8; ++r)
              ldsP[(r + 8 * hi) * 32 + jj * 16 + l16] = (__bf16)sacc[j][r];
          }
          AF pf;
          #pragma unroll
          for (int p = 0; p < 8; ++p) {
            int col = 2 * p + 16 * hi;
            pf.u[p] = *(const unsigned*)(ldsP + l16 * 32 + col);
          }
          #pragma unroll
          for (int n = 0; n < 8; ++n) {
            const v16bf bv = *(const v16bf*)(ldsV + (n * 16 + l16) * KSTRIDE + jp * 32 + 16 * hi);
            oacc[n] = __builtin_amdgcn_wmma_f32_16x16x32_bf16(
                false, pf.v, false, bv, (short)0, oacc[n], false, false);
          }
        }
      }
      __syncthreads();
    }
  }

  // ---- store O (16 q-rows x 128 d per wave), streaming NT ----
  float* orow = Out + ((size_t)b * NN + qbase) * DD;
  #pragma unroll
  for (int n = 0; n < 8; ++n) {
    #pragma unroll
    for (int r = 0; r < 8; ++r) {
      int q = r + 8 * hi;
      int d = n * 16 + l16;
      __builtin_nontemporal_store(oacc[n][r], &orow[(size_t)q * DD + d]);
    }
  }
}

extern "C" void kernel_launch(void* const* d_in, const int* in_sizes, int n_in,
                              void* d_out, int out_size, void* d_ws, size_t ws_size,
                              hipStream_t stream) {
  (void)in_sizes; (void)n_in; (void)d_ws; (void)ws_size; (void)out_size;
  const float* Q = (const float*)d_in[0];
  const float* K = (const float*)d_in[1];
  const float* V = (const float*)d_in[2];
  float* Out = (float*)d_out;                                   // [B,N,D]
  float* Att = (float*)d_out + (size_t)BATCH * NN * DD;         // [B,N,N]

  dim3 grid(NN / 128, BATCH);
  dim3 block(256);
  size_t shmem = (size_t)(2 * 128 * KSTRIDE + 8 * 16 * 32) * sizeof(__bf16); // 81920 B
  attn_fused_kernel<<<grid, block, shmem, stream>>>(Q, K, V, Out, Att);
}
